// TransformerBlock_41936060678261
// MI455X (gfx1250) — matrix operand order repaired and checked by reading, not compiled, not run on hardware
//
#include <hip/hip_runtime.h>

// ---------------------------------------------------------------- types
typedef __bf16 bf16_t;
typedef __bf16 v16bf __attribute__((ext_vector_type(16)));
typedef float  v8f   __attribute__((ext_vector_type(8)));
typedef unsigned v4u __attribute__((ext_vector_type(4)));
typedef int      v8i __attribute__((ext_vector_type(8)));
typedef int      v4i __attribute__((ext_vector_type(4)));

static __device__ inline bf16_t f2bf(float f) {
    unsigned u = __builtin_bit_cast(unsigned, f);
    u += 0x7FFFu + ((u >> 16) & 1u);          // round-to-nearest-even
    unsigned short h = (unsigned short)(u >> 16);
    return __builtin_bit_cast(bf16_t, h);
}
static __device__ inline float bf2f(bf16_t b) {
    unsigned short h = __builtin_bit_cast(unsigned short, b);
    unsigned u = ((unsigned)h) << 16;
    return __builtin_bit_cast(float, u);
}
static __device__ inline unsigned pack2bf(float a, float b) {
    return (unsigned)__builtin_bit_cast(unsigned short, f2bf(a)) |
           ((unsigned)__builtin_bit_cast(unsigned short, f2bf(b)) << 16);
}
static __device__ inline v8f wmma_bf16(v16bf a, v16bf b, v8f c) {
    return __builtin_amdgcn_wmma_f32_16x16x32_bf16(
        false, a, false, b, (short)0, c, false, false);
}

union frag16 { uint4 q[2]; v16bf v; };
// A-frag layout: 16 contiguous K elements starting at 16*half
static __device__ inline v16bf load_frag_a(const bf16_t* base, int half) {
    frag16 f;
    f.q[0] = *(const uint4*)(base + 16 * half);
    f.q[1] = *(const uint4*)(base + 16 * half + 8);
    return f.v;
}
// B-frag layout: 16 contiguous K elements starting at 16*half
static __device__ inline v16bf load_frag_b(const bf16_t* base, int half) {
    frag16 f;
    f.q[0] = *(const uint4*)(base + 16 * half);
    f.q[1] = *(const uint4*)(base + 16 * half + 8);
    return f.v;
}

#define NTOK 2048   // B*T
#define DMODEL 1024

// ---------------------------------------------------------------- rmsnorm -> bf16
__global__ __launch_bounds__(256)
void rmsnorm_kernel(const float* __restrict__ x, const float* __restrict__ w,
                    bf16_t* __restrict__ out) {
    __shared__ float red[8];
    const int row = blockIdx.x;
    const float* xr = x + (size_t)row * DMODEL;
    float ss = 0.f;
    for (int i = threadIdx.x; i < DMODEL; i += 256) { float v = xr[i]; ss += v * v; }
    for (int m = 1; m < 32; m <<= 1) ss += __shfl_xor(ss, m, 32);
    if ((threadIdx.x & 31) == 0) red[threadIdx.x >> 5] = ss;
    __syncthreads();
    float tot = 0.f;
#pragma unroll
    for (int i = 0; i < 8; ++i) tot += red[i];
    const float rinv = rsqrtf(tot / (float)DMODEL + 1e-6f);
    for (int i = threadIdx.x; i < DMODEL; i += 256)
        out[(size_t)row * DMODEL + i] = f2bf(xr[i] * rinv * w[i]);
}

// ---------------------------------------------------------------- generic bf16 WMMA GEMM
// C[M x Ncols] = A_bf16[M x K] * B_f32[K x Ncols]
// B tile fetched global->LDS by the Tensor Data Mover, converted to bf16
// transposed (fragment-friendly) in LDS. epi: 0=store  2=store+resid  3=accumulate
__global__ __launch_bounds__(256)
void gemm_bf16(const bf16_t* __restrict__ A, const float* __restrict__ Bm,
               float* __restrict__ C, const float* __restrict__ resid,
               int M, int Ncols, int K, int epi) {
    __shared__ bf16_t As[64 * 32];        //  4 KB
    __shared__ float  Bsf[32 * 128];      // 16 KB TDM landing pad
    __shared__ bf16_t BsT[128 * 40];      // 10 KB transposed bf16 tile (padded)
    const int tid  = threadIdx.x;
    const int lane = tid & 31;
    const int w    = tid >> 5;
    const int wm   = w & 3, wn = w >> 2;  // 4 waves along M, 2 along N
    const int row0 = blockIdx.x * 64;
    const int col0 = blockIdx.y * 128;
    const int n16  = lane & 15, half = lane >> 4;

    const unsigned lds_bsf = (unsigned)(size_t)(void*)Bsf;  // LDS byte offset

    v8f acc[4] = {};
    for (int k0 = 0; k0 < K; k0 += 32) {
        // A tile: 256 threads x 1 uint4 (8 bf16 each)
        {
            int r = tid >> 2, c8 = tid & 3;
            *(uint4*)&As[r * 32 + 8 * c8] =
                *(const uint4*)&A[(size_t)(row0 + r) * K + k0 + 8 * c8];
        }
        // B tile: TDM async DMA of 32x128 f32 slab, issued by wave 0
        if (w == 0) {
            unsigned long long ga =
                (unsigned long long)(const void*)(Bm + (size_t)k0 * Ncols + col0);
            v4u g0;
            g0[0] = 1u;                                   // count=1, user mode
            g0[1] = lds_bsf;                              // lds_addr
            g0[2] = (unsigned)(ga & 0xFFFFFFFFu);         // global_addr[31:0]
            g0[3] = (unsigned)((ga >> 32) & 0x1FFFFFFu) | (2u << 30); // addr[56:32], type=2
            unsigned d0 = (unsigned)(Ncols - col0);       // tensor_dim0
            unsigned d1 = (unsigned)(K - k0);             // tensor_dim1
            v8i g1;
            g1[0] = (int)(2u << 16);                      // data_size=4B, mask=0
            g1[1] = (int)((d0 & 0xFFFFu) << 16);          // dim0[15:0] @ bits63:48
            g1[2] = (int)((d0 >> 16) | ((d1 & 0xFFFFu) << 16));
            g1[3] = (int)((d1 >> 16) | (128u << 16));     // tile_dim0=128
            g1[4] = 32;                                   // tile_dim1=32, tile_dim2=0
            g1[5] = Ncols;                                // tensor_dim0_stride lo32
            g1[6] = 0;
            g1[7] = 0;
            v4i gz4 = {0, 0, 0, 0};
            v8i gz8 = {0, 0, 0, 0, 0, 0, 0, 0};
            __builtin_amdgcn_tensor_load_to_lds(g0, g1, gz4, gz4, gz8, 0);
            __builtin_amdgcn_s_wait_tensorcnt(0);
        }
        __syncthreads();
        // convert f32 slab -> bf16 transposed [col][k], pack 2 k per dword
        for (int i = tid; i < 2048; i += 256) {
            int c = i & 127, rp = i >> 7;
            float f0 = Bsf[(2 * rp) * 128 + c];
            float f1 = Bsf[(2 * rp + 1) * 128 + c];
            *(unsigned*)&BsT[c * 40 + 2 * rp] = pack2bf(f0, f1);
        }
        __syncthreads();

        v16bf a = load_frag_a(&As[(16 * wm + n16) * 32], half);
#pragma unroll
        for (int ns = 0; ns < 4; ++ns) {
            v16bf b = load_frag_b(&BsT[(64 * wn + 16 * ns + n16) * 40], half);
            acc[ns] = wmma_bf16(a, b, acc[ns]);
        }
        __syncthreads();
    }
#pragma unroll
    for (int ns = 0; ns < 4; ++ns)
#pragma unroll
        for (int r = 0; r < 8; ++r) {
            int row = row0 + 16 * wm + r + 8 * half;
            int col = col0 + 64 * wn + 16 * ns + n16;
            size_t idx = (size_t)row * Ncols + col;
            float v = acc[ns][r];
            if (epi == 0)       C[idx] = v;
            else if (epi == 2)  C[idx] = v + resid[idx];
            else                C[idx] += v;
        }
}

// ---------------------------------------------------------------- RoPE + bf16 pack for Q,K,V
__global__ void rope_pack_kernel(const float* __restrict__ qf, const float* __restrict__ kf,
                                 const float* __restrict__ vf, const float* __restrict__ fcos,
                                 const float* __restrict__ fsin, bf16_t* __restrict__ qb,
                                 bf16_t* __restrict__ kb, bf16_t* __restrict__ vb) {
    const int QW = NTOK * 16 * 32, KW = NTOK * 4 * 32, VW = NTOK * 256;
    for (int idx = blockIdx.x * blockDim.x + threadIdx.x; idx < QW + KW + VW;
         idx += gridDim.x * blockDim.x) {
        if (idx < QW) {                       // Q: 16 heads, fold in HD^-0.5 = 1/8
            int p = idx;
            int i = p & 31;  p >>= 5;
            int hh = p & 15; p >>= 4;
            int tok = p, t = tok & 1023;
            float c = fcos[t * 32 + i], s = fsin[t * 32 + i];
            size_t base = (size_t)tok * DMODEL + hh * 64 + 2 * i;
            float a = qf[base], b = qf[base + 1];
            *(unsigned*)(qb + base) = pack2bf((a * c - b * s) * 0.125f,
                                              (a * s + b * c) * 0.125f);
        } else if (idx < QW + KW) {           // K: 4 kv heads
            int p = idx - QW;
            int i = p & 31; p >>= 5;
            int hh = p & 3; p >>= 2;
            int tok = p, t = tok & 1023;
            float c = fcos[t * 32 + i], s = fsin[t * 32 + i];
            size_t base = (size_t)tok * 256 + hh * 64 + 2 * i;
            float a = kf[base], b = kf[base + 1];
            *(unsigned*)(kb + base) = pack2bf(a * c - b * s, a * s + b * c);
        } else {                              // V: straight convert
            int p = idx - QW - KW;
            vb[p] = f2bf(vf[p]);
        }
    }
}

// ---------------------------------------------------------------- causal flash attention (GQA 4:1)
// grid (T/64, H, B), 128 threads = 4 waves, each wave owns 16 query rows.
__global__ __launch_bounds__(128)
void attn_kernel(const bf16_t* __restrict__ qb, const bf16_t* __restrict__ kb,
                 const bf16_t* __restrict__ vb, bf16_t* __restrict__ ob) {
    __shared__ bf16_t Kt[32 * 64];     // [key][hd]   (K-frag contiguous in hd)
    __shared__ bf16_t VtT[64 * 40];    // [hd][key]   (V-frag contiguous in key)
    __shared__ bf16_t Pw[4][16 * 32];
    const int lane = threadIdx.x & 31;
    const int w    = threadIdx.x >> 5;
    const int q0   = blockIdx.x * 64;
    const int h    = blockIdx.y;
    const int b    = blockIdx.z;
    const int kvh  = h >> 2;
    const int n16  = lane & 15, half = lane >> 4;
    const int qw   = q0 + 16 * w;
    const int tok0 = b * 1024;

    v16bf aq[2];
#pragma unroll
    for (int j = 0; j < 2; ++j)
        aq[j] = load_frag_a(qb + (size_t)(tok0 + qw + n16) * DMODEL + h * 64 + 32 * j, half);

    v8f o[4] = {};
    float mrow[8], lrow[8];
#pragma unroll
    for (int r = 0; r < 8; ++r) { mrow[r] = -1e30f; lrow[r] = 0.f; }

    const int ktiles = (q0 + 64) >> 5;
    for (int kt = 0; kt < ktiles; ++kt) {
        const int k0 = kt * 32;
        for (int i = threadIdx.x; i < 1024; i += 128) {
            int kl = i >> 5, hd = (i & 31) * 2;
            size_t src = (size_t)(tok0 + k0 + kl) * 256 + kvh * 64 + hd;
            *(unsigned*)&Kt[kl * 64 + hd] = *(const unsigned*)(kb + src);
            unsigned vv = *(const unsigned*)(vb + src);
            VtT[hd * 40 + kl]       = __builtin_bit_cast(bf16_t, (unsigned short)(vv & 0xFFFFu));
            VtT[(hd + 1) * 40 + kl] = __builtin_bit_cast(bf16_t, (unsigned short)(vv >> 16));
        }
        __syncthreads();
        if (k0 <= qw + 15) {
            v8f s[2]; s[0] = {}; s[1] = {};
#pragma unroll
            for (int st = 0; st < 2; ++st)
#pragma unroll
                for (int j = 0; j < 2; ++j) {
                    v16bf bk = load_frag_b(&Kt[(16 * st + n16) * 64 + 32 * j], half);
                    s[st] = wmma_bf16(aq[j], bk, s[st]);
                }
            float pn0[8], pn1[8];
#pragma unroll
            for (int r = 0; r < 8; ++r) {
                int qrow = qw + r + 8 * half;
                float s0v = s[0][r] + (((k0 + n16)      > qrow) ? -1e30f : 0.f);
                float s1v = s[1][r] + (((k0 + 16 + n16) > qrow) ? -1e30f : 0.f);
                float mx = fmaxf(s0v, s1v);
                for (int msk = 1; msk < 16; msk <<= 1) mx = fmaxf(mx, __shfl_xor(mx, msk, 32));
                float mnew  = fmaxf(mrow[r], mx);
                float scale = __expf(mrow[r] - mnew);
                float p0 = __expf(s0v - mnew), p1 = __expf(s1v - mnew);
                float ps = p0 + p1;
                for (int msk = 1; msk < 16; msk <<= 1) ps += __shfl_xor(ps, msk, 32);
                lrow[r] = lrow[r] * scale + ps;
                mrow[r] = mnew;
#pragma unroll
                for (int f = 0; f < 4; ++f) o[f][r] *= scale;
                pn0[r] = p0; pn1[r] = p1;
            }
            // store P (C-layout -> row-major) for the A-fragment reload
#pragma unroll
            for (int r = 0; r < 8; ++r) {
                Pw[w][(r + 8 * half) * 32 + n16]      = f2bf(pn0[r]);
                Pw[w][(r + 8 * half) * 32 + 16 + n16] = f2bf(pn1[r]);
            }
            v16bf ap = load_frag_a(&Pw[w][n16 * 32], half);
#pragma unroll
            for (int f = 0; f < 4; ++f) {
                v16bf bv = load_frag_b(&VtT[(16 * f + n16) * 40], half);
                o[f] = wmma_bf16(ap, bv, o[f]);
            }
        }
        __syncthreads();
    }
#pragma unroll
    for (int r = 0; r < 8; ++r) {
        int qrow = qw + r + 8 * half;
        size_t base = (size_t)(tok0 + qrow) * DMODEL + h * 64;
        float inv = 1.0f / lrow[r];
#pragma unroll
        for (int f = 0; f < 4; ++f)
            ob[base + 16 * f + n16] = f2bf(o[f][r] * inv);
    }
}

// ---------------------------------------------------------------- router + softmax + top-4
__global__ __launch_bounds__(32)
void router_topk_kernel(const bf16_t* __restrict__ h2, const float* __restrict__ rw,
                        float* __restrict__ combine, int* __restrict__ counts,
                        int* __restrict__ lists) {
    __shared__ float sc[32];
    const int tok = blockIdx.x;
    const int e   = threadIdx.x;
    float acc = 0.f;
    for (int d = 0; d < DMODEL; ++d)
        acc += bf2f(h2[(size_t)tok * DMODEL + d]) * rw[d * 32 + e];
    float mx = acc;
    for (int m = 1; m < 32; m <<= 1) mx = fmaxf(mx, __shfl_xor(mx, m, 32));
    float ex = __expf(acc - mx);
    float sm = ex;
    for (int m = 1; m < 32; m <<= 1) sm += __shfl_xor(sm, m, 32);
    sc[e] = ex / sm;
    combine[(size_t)tok * 32 + e] = 0.f;
    __syncthreads();
    if (e == 0) {
#pragma unroll
        for (int t = 0; t < 4; ++t) {
            int best = 0; float bv = -1.f;
            for (int j = 0; j < 32; ++j) if (sc[j] > bv) { bv = sc[j]; best = j; }
            sc[best] = -2.f;
            combine[(size_t)tok * 32 + best] = bv;
            int pos = atomicAdd(&counts[best], 1);
            lists[best * NTOK + pos] = tok;
        }
    }
}

// ---------------------------------------------------------------- routed experts (gather + WMMA)
__global__ __launch_bounds__(256)
void moe_routed_kernel(const bf16_t* __restrict__ h2, const float* __restrict__ wg,
                       const float* __restrict__ wu, const float* __restrict__ wd,
                       const float* __restrict__ combine, const int* __restrict__ counts,
                       const int* __restrict__ lists, float* __restrict__ out) {
    __shared__ bf16_t Ah[16 * DMODEL];   // 32 KB gathered activations
    __shared__ bf16_t Hh[16 * 256];      // 8 KB gated intermediate
    __shared__ float  cw[16];
    __shared__ int    toks[16];

    const int e    = blockIdx.x;
    const int cnt  = counts[e];
    const int base = blockIdx.y * 16;
    if (base >= cnt) return;
    const int valid = (cnt - base < 16) ? (cnt - base) : 16;

    const int tid  = threadIdx.x;
    const int lane = tid & 31, w = tid >> 5;
    const int n16  = lane & 15, half = lane >> 4;

    if (tid < 16) {
        int tk = (tid < valid) ? lists[e * NTOK + base + tid] : 0;
        toks[tid] = tk;
        cw[tid]   = (tid < valid) ? combine[(size_t)tk * 32 + e] : 0.f;
    }
    __syncthreads();
    for (int i = tid; i < 2048; i += 256) {          // 16 rows x 128 uint4
        int r = i >> 7, d8 = (i & 127) * 8;
        *(uint4*)&Ah[r * DMODEL + d8] =
            *(const uint4*)&h2[(size_t)toks[r] * DMODEL + d8];
    }
    __syncthreads();

    // gate & up: [16x1024] @ [1024x256]; wave w owns cols 32w..32w+31
    v8f ag[2] = {}, au[2] = {};
    const size_t wbase = (size_t)e * DMODEL * 256;
    for (int k0 = 0; k0 < DMODEL; k0 += 32) {
        v16bf a = load_frag_a(&Ah[n16 * DMODEL + k0], half);
#pragma unroll
        for (int ns = 0; ns < 2; ++ns) {
            int col = 32 * w + 16 * ns + n16;
            v16bf bg, bu;
#pragma unroll
            for (int i = 0; i < 16; ++i) {
                int kk = k0 + 16 * half + i;
                bg[i] = f2bf(wg[wbase + (size_t)kk * 256 + col]);
                bu[i] = f2bf(wu[wbase + (size_t)kk * 256 + col]);
            }
            ag[ns] = wmma_bf16(a, bg, ag[ns]);
            au[ns] = wmma_bf16(a, bu, au[ns]);
        }
    }
#pragma unroll
    for (int ns = 0; ns < 2; ++ns)
#pragma unroll
        for (int r = 0; r < 8; ++r) {
            int m = r + 8 * half;
            float gv = ag[ns][r];
            float hv = (gv / (1.f + __expf(-gv))) * au[ns][r] * cw[m];
            Hh[m * 256 + 32 * w + 16 * ns + n16] = f2bf(hv);
        }
    __syncthreads();

    // down: [16x256] @ [256x1024]; wave w owns cols 128w..128w+127
    v8f od[8] = {};
    const size_t dbase = (size_t)e * 256 * DMODEL;
    for (int k0 = 0; k0 < 256; k0 += 32) {
        v16bf a = load_frag_a(&Hh[n16 * 256 + k0], half);
#pragma unroll
        for (int ns = 0; ns < 8; ++ns) {
            int col = 128 * w + 16 * ns + n16;
            v16bf bd;
#pragma unroll
            for (int i = 0; i < 16; ++i) {
                int kk = k0 + 16 * half + i;
                bd[i] = f2bf(wd[dbase + (size_t)kk * DMODEL + col]);
            }
            od[ns] = wmma_bf16(a, bd, od[ns]);
        }
    }
#pragma unroll
    for (int ns = 0; ns < 8; ++ns)
#pragma unroll
        for (int r = 0; r < 8; ++r) {
            int m = r + 8 * half;
            if (m < valid) {
                int col = 128 * w + 16 * ns + n16;
                atomicAdd(&out[(size_t)toks[m] * DMODEL + col], od[ns][r]);
            }
        }
}

// ---------------------------------------------------------------- silu(g)*u -> bf16
__global__ void silu_mul_kernel(const float* __restrict__ g, const float* __restrict__ u,
                                bf16_t* __restrict__ s, int n) {
    for (int i = blockIdx.x * blockDim.x + threadIdx.x; i < n; i += gridDim.x * blockDim.x) {
        float gv = g[i];
        s[i] = f2bf((gv / (1.f + __expf(-gv))) * u[i]);
    }
}

// ---------------------------------------------------------------- launch
extern "C" void kernel_launch(void* const* d_in, const int* in_sizes, int n_in,
                              void* d_out, int out_size, void* d_ws, size_t ws_size,
                              hipStream_t stream) {
    const float* x    = (const float*)d_in[0];
    const float* fcos = (const float*)d_in[1];
    const float* fsin = (const float*)d_in[2];
    const float* n1w  = (const float*)d_in[3];
    const float* n2w  = (const float*)d_in[4];
    const float* wq   = (const float*)d_in[5];
    const float* wk   = (const float*)d_in[6];
    const float* wv   = (const float*)d_in[7];
    const float* wo   = (const float*)d_in[8];
    const float* rw   = (const float*)d_in[9];
    const float* wg   = (const float*)d_in[10];
    const float* wu   = (const float*)d_in[11];
    const float* wd   = (const float*)d_in[12];
    const float* shg  = (const float*)d_in[13];
    const float* shu  = (const float*)d_in[14];
    const float* shd  = (const float*)d_in[15];
    float* out = (float*)d_out;

    char* ws = (char*)d_ws;
    size_t off = 0;
    auto alloc = [&](size_t b) -> void* {
        void* p = (void*)(ws + off);
        off += (b + 255) & ~((size_t)255);
        return p;
    };
    const size_t N = NTOK, D = DMODEL;
    bf16_t* h1b  = (bf16_t*)alloc(N * D * 2);
    float*  qf   = (float*) alloc(N * D * 4);
    float*  kf   = (float*) alloc(N * 256 * 4);
    float*  vf   = (float*) alloc(N * 256 * 4);
    bf16_t* qb   = (bf16_t*)alloc(N * D * 2);
    bf16_t* kb   = (bf16_t*)alloc(N * 256 * 2);
    bf16_t* vb   = (bf16_t*)alloc(N * 256 * 2);
    bf16_t* ab   = (bf16_t*)alloc(N * D * 2);
    float*  x2   = (float*) alloc(N * D * 4);
    bf16_t* h2b  = (bf16_t*)alloc(N * D * 2);
    float*  gf   = (float*) alloc(N * 512 * 4);
    float*  uf   = (float*) alloc(N * 512 * 4);
    bf16_t* sb   = (bf16_t*)alloc(N * 512 * 2);
    float*  comb = (float*) alloc(N * 32 * 4);
    int*    cnts = (int*)   alloc(32 * 4);
    int*    lst  = (int*)   alloc(32 * N * 4);

    rmsnorm_kernel<<<NTOK, 256, 0, stream>>>(x, n1w, h1b);
    gemm_bf16<<<dim3(NTOK / 64, 1024 / 128), 256, 0, stream>>>(h1b, wq, qf, nullptr, NTOK, 1024, 1024, 0);
    gemm_bf16<<<dim3(NTOK / 64,  256 / 128), 256, 0, stream>>>(h1b, wk, kf, nullptr, NTOK,  256, 1024, 0);
    gemm_bf16<<<dim3(NTOK / 64,  256 / 128), 256, 0, stream>>>(h1b, wv, vf, nullptr, NTOK,  256, 1024, 0);
    rope_pack_kernel<<<2048, 256, 0, stream>>>(qf, kf, vf, fcos, fsin, qb, kb, vb);
    attn_kernel<<<dim3(16, 16, 2), 128, 0, stream>>>(qb, kb, vb, ab);
    gemm_bf16<<<dim3(NTOK / 64, 1024 / 128), 256, 0, stream>>>(ab, wo, x2, x, NTOK, 1024, 1024, 2);
    rmsnorm_kernel<<<NTOK, 256, 0, stream>>>(x2, n2w, h2b);
    (void)hipMemsetAsync(cnts, 0, 32 * sizeof(int), stream);
    router_topk_kernel<<<NTOK, 32, 0, stream>>>(h2b, rw, comb, cnts, lst);
    (void)hipMemcpyAsync(out, x2, N * D * sizeof(float), hipMemcpyDeviceToDevice, stream);
    moe_routed_kernel<<<dim3(32, NTOK / 16), 256, 0, stream>>>(h2b, wg, wu, wd, comb, cnts, lst, out);
    gemm_bf16<<<dim3(NTOK / 64, 512 / 128), 256, 0, stream>>>(h2b, shg, gf, nullptr, NTOK, 512, 1024, 0);
    gemm_bf16<<<dim3(NTOK / 64, 512 / 128), 256, 0, stream>>>(h2b, shu, uf, nullptr, NTOK, 512, 1024, 0);
    silu_mul_kernel<<<2048, 256, 0, stream>>>(gf, uf, sb, (int)(N * 512));
    gemm_bf16<<<dim3(NTOK / 64, 1024 / 128), 256, 0, stream>>>(sb, shd, out, nullptr, NTOK, 1024, 512, 3);
}
